// Head_84310208021168
// MI455X (gfx1250) — matrix operand order repaired and checked by reading, not compiled, not run on hardware
//
#include <hip/hip_runtime.h>

typedef __attribute__((ext_vector_type(16))) _Float16 v16h;
typedef __attribute__((ext_vector_type(8)))  float    v8f;

constexpr int BATCH = 8;
constexpr int SEQ   = 4096;
constexpr int NE    = 512;   // n_embed (reduction dim for projections, softmax scale base)
constexpr int HS    = 64;    // head size
constexpr int ILEN  = 1024;  // prefix length (fully visible block)

__device__ inline v8f zero8() {
  v8f z;
  #pragma unroll
  for (int i = 0; i < 8; ++i) z[i] = 0.f;
  return z;
}

__device__ inline v16h load16h(const _Float16* p) {
  union { uint4 u[2]; v16h h; } t;
  t.u[0] = *(const uint4*)(p);
  t.u[1] = *(const uint4*)(p + 8);
  return t.h;
}

// ---------------------------------------------------------------------------
// Kernel 0: convert the three f32 weight matrices [NE, HS] into f16,
// transposed to [HS, NE] so WMMA B-fragments read contiguous memory.
// ---------------------------------------------------------------------------
__global__ void convert_weights(const float* __restrict__ Wq,
                                const float* __restrict__ Wk,
                                const float* __restrict__ Wv,
                                _Float16* __restrict__ WqT,
                                _Float16* __restrict__ WkT,
                                _Float16* __restrict__ WvT) {
  int idx = blockIdx.x * blockDim.x + threadIdx.x;
  if (idx >= HS * NE) return;
  int h = idx / NE;
  int c = idx % NE;
  WqT[h * NE + c] = (_Float16)Wq[c * HS + h];
  WkT[h * NE + c] = (_Float16)Wk[c * HS + h];
  WvT[h * NE + c] = (_Float16)Wv[c * HS + h];
}

// ---------------------------------------------------------------------------
// Kernel 1: q/k/v projections. One wave (32 lanes) per 16-row tile.
// Accumulates in f32 via v_wmma_f32_16x16x32_f16, K = NE = 512 (16 steps).
// q stored with 512^-0.5 * log2(e) folded in (softmax runs in log2 domain).
// v stored transposed [B, HS, SEQ] for contiguous P@V B-fragments.
// ---------------------------------------------------------------------------
__global__ __launch_bounds__(32)
void proj_qkv(const float* __restrict__ x,
              const _Float16* __restrict__ WqT,
              const _Float16* __restrict__ WkT,
              const _Float16* __restrict__ WvT,
              _Float16* __restrict__ qh,   // [B, SEQ, HS]  (pre-scaled)
              _Float16* __restrict__ kh,   // [B, SEQ, HS]
              _Float16* __restrict__ vT) { // [B, HS, SEQ]
  const int lane = threadIdx.x & 31;
  const int row  = lane & 15;     // M (A) / N (B,C)
  const int hi   = lane >> 4;     // lane-half selector
  const int tile = blockIdx.x;
  const int b    = tile / (SEQ / 16);
  const int t0   = (tile % (SEQ / 16)) * 16;

  const float* xrow = x + ((size_t)b * SEQ + t0 + row) * NE;
  const _Float16* Wt[3] = {WqT, WkT, WvT};

  v8f acc[3][4];
  #pragma unroll
  for (int m = 0; m < 3; ++m)
    #pragma unroll
    for (int n = 0; n < 4; ++n) acc[m][n] = zero8();

  for (int kk = 0; kk < NE; kk += 32) {
    if (kk + 64 < NE) __builtin_prefetch(xrow + kk + 64, 0, 1);  // global_prefetch_b8
    // A fragment (16x32 f16): lane holds row `row`, K = hi*16 + j.
    v16h a;
    const int abase = kk + hi * 16;
    #pragma unroll
    for (int vg = 0; vg < 8; ++vg) {
      const int k = abase + 2 * vg;
      a[2 * vg + 0] = (_Float16)xrow[k + 0];
      a[2 * vg + 1] = (_Float16)xrow[k + 1];
    }
    // B fragments (32x16 f16): lane holds column n*16+row, K = hi*16 + j.
    #pragma unroll
    for (int m = 0; m < 3; ++m) {
      #pragma unroll
      for (int n = 0; n < 4; ++n) {
        const _Float16* wp = Wt[m] + (size_t)(n * 16 + row) * NE + kk + hi * 16;
        v16h bf = load16h(wp);
        acc[m][n] = __builtin_amdgcn_wmma_f32_16x16x32_f16(
            false, a, false, bf, (short)0, acc[m][n], false, false);
      }
    }
  }

  // softmax scale (n_embed^-0.5) * log2(e): logits end up in log2 domain.
  const float qscale = rsqrtf((float)NE) * 1.44269504088896f;
  #pragma unroll
  for (int r = 0; r < 8; ++r) {
    const int M = hi ? (8 + r) : r;                 // C-layout row
    const size_t trow = (size_t)b * SEQ + t0 + M;
    #pragma unroll
    for (int n = 0; n < 4; ++n) {
      const int h = n * 16 + row;                   // C-layout column
      qh[trow * HS + h] = (_Float16)(acc[0][n][r] * qscale);
      kh[trow * HS + h] = (_Float16)(acc[1][n][r]);
      vT[((size_t)b * HS + h) * SEQ + t0 + M] = (_Float16)(acc[2][n][r]);
    }
  }
}

// ---------------------------------------------------------------------------
// Kernel 2: flash attention with prefix-causal mask.
// One wave per 16-query tile; 32 keys per step (2 S-tiles of 16x16).
// visible(t, s) = (s < ILEN) || (s <= t).  Softmax in log2 domain (exp2).
// Software pipelined: next-step K frags load during softmax VALU,
// next-step V frags load during next S-WMMA/mask stretch.
// ---------------------------------------------------------------------------
__global__ __launch_bounds__(32)
void attn(const _Float16* __restrict__ qh,
          const _Float16* __restrict__ kh,
          const _Float16* __restrict__ vT,
          float* __restrict__ out) {      // [B, SEQ, HS] f32
  __shared__ _Float16 Qlds[16][64];
  __shared__ _Float16 Plds[16][32];

  const int lane = threadIdx.x & 31;
  const int row  = lane & 15;
  const int hi   = lane >> 4;
  const int tile = blockIdx.x;
  const int b    = tile / (SEQ / 16);
  const int t0   = (tile % (SEQ / 16)) * 16;

  const _Float16* kbase = kh + (size_t)b * SEQ * HS;
  const _Float16* vbase = vT + (size_t)b * HS * SEQ;

  // Cooperative load of the 16x64 f16 Q tile (each lane: half a row).
  {
    const int qr = lane >> 1, qo = (lane & 1) * 32;
    const _Float16* src = qh + ((size_t)b * SEQ + t0 + qr) * HS + qo;
    uint4 u0 = *(const uint4*)(src + 0);
    uint4 u1 = *(const uint4*)(src + 8);
    uint4 u2 = *(const uint4*)(src + 16);
    uint4 u3 = *(const uint4*)(src + 24);
    *(uint4*)&Qlds[qr][qo + 0]  = u0;
    *(uint4*)&Qlds[qr][qo + 8]  = u1;
    *(uint4*)&Qlds[qr][qo + 16] = u2;
    *(uint4*)&Qlds[qr][qo + 24] = u3;
  }
  __syncthreads();

  // Q A-fragments for h = 0..31 (qa) and h = 32..63 (qb).
  v16h qa, qb;
  #pragma unroll
  for (int vg = 0; vg < 8; ++vg) {
    const int kA = hi * 16 + 2 * vg;
    qa[2 * vg + 0] = Qlds[row][kA + 0];
    qa[2 * vg + 1] = Qlds[row][kA + 1];
    qb[2 * vg + 0] = Qlds[row][32 + kA + 0];
    qb[2 * vg + 1] = Qlds[row][32 + kA + 1];
  }

  v8f o[4];
  #pragma unroll
  for (int n = 0; n < 4; ++n) o[n] = zero8();
  float m_i[8], l_i[8];
  #pragma unroll
  for (int r = 0; r < 8; ++r) { m_i[r] = -1e30f; l_i[r] = 0.f; }

  const int limit  = min(SEQ, max(ILEN, t0 + 16));
  const int nsteps = (limit + 31) / 32;

  // Preload step-0 fragments.
  v16h kf[4], vf[4];
  {
    const _Float16* kp0 = kbase + (size_t)(0 + row) * HS + hi * 16;
    kf[0] = load16h(kp0);
    kf[1] = load16h(kp0 + 32);
    kf[2] = load16h(kp0 + 16 * HS);
    kf[3] = load16h(kp0 + 16 * HS + 32);
    #pragma unroll
    for (int n = 0; n < 4; ++n)
      vf[n] = load16h(vbase + (size_t)(n * 16 + row) * SEQ + hi * 16);
  }

  for (int st = 0; st < nsteps; ++st) {
    const int s0 = st * 32;
    const int s1 = s0 + 32;                 // next step
    const bool more = (st + 1) < nsteps;

    // S = (scaled q) @ k^T for 32 keys: two 16x16 tiles.
    v8f sA = zero8(), sB = zero8();
    sA = __builtin_amdgcn_wmma_f32_16x16x32_f16(false, qa, false, kf[0], (short)0, sA, false, false);
    sA = __builtin_amdgcn_wmma_f32_16x16x32_f16(false, qb, false, kf[1], (short)0, sA, false, false);
    sB = __builtin_amdgcn_wmma_f32_16x16x32_f16(false, qa, false, kf[2], (short)0, sB, false, false);
    sB = __builtin_amdgcn_wmma_f32_16x16x32_f16(false, qb, false, kf[3], (short)0, sB, false, false);

    // Issue next-step K fragment loads; they complete during the softmax VALU.
    if (more) {
      const _Float16* kp0 = kbase + (size_t)(s1 + row) * HS + hi * 16;
      kf[0] = load16h(kp0);
      kf[1] = load16h(kp0 + 32);
      kf[2] = load16h(kp0 + 16 * HS);
      kf[3] = load16h(kp0 + 16 * HS + 32);
      if (s1 + 32 < limit)  // pull the step-after-next K tile toward L2/L0
        __builtin_prefetch(kbase + (size_t)(s1 + 32 + lane) * HS, 0, 1);
    }

    // Online softmax (log2 domain) per query row; rows live across 16 lanes.
    #pragma unroll
    for (int r = 0; r < 8; ++r) {
      const int tg = t0 + (hi ? (8 + r) : r);
      const int sa = s0 + row;
      const int sb = sa + 16;
      float vA = ((sa < ILEN) || (sa <= tg)) ? sA[r] : -1e30f;
      float vB = ((sb < ILEN) || (sb <= tg)) ? sB[r] : -1e30f;
      float mx = fmaxf(vA, vB);
      mx = fmaxf(mx, __shfl_xor(mx, 1, 32));
      mx = fmaxf(mx, __shfl_xor(mx, 2, 32));
      mx = fmaxf(mx, __shfl_xor(mx, 4, 32));
      mx = fmaxf(mx, __shfl_xor(mx, 8, 32));
      const float mnew = fmaxf(m_i[r], mx);
      const float corr = exp2f(m_i[r] - mnew);
      const float pA = exp2f(vA - mnew);
      const float pB = exp2f(vB - mnew);
      float ps = pA + pB;
      ps += __shfl_xor(ps, 1, 32);
      ps += __shfl_xor(ps, 2, 32);
      ps += __shfl_xor(ps, 4, 32);
      ps += __shfl_xor(ps, 8, 32);
      l_i[r] = l_i[r] * corr + ps;
      m_i[r] = mnew;
      #pragma unroll
      for (int n = 0; n < 4; ++n) o[n][r] *= corr;   // same row mapping as S
      const int rw = hi ? (8 + r) : r;
      Plds[rw][row]      = (_Float16)pA;
      Plds[rw][row + 16] = (_Float16)pB;
    }
    __syncthreads();

    // P A-fragment (16x32) from LDS (C-layout -> A-layout transpose).
    v16h pa;
    #pragma unroll
    for (int vg = 0; vg < 8; ++vg) {
      const int kA = hi * 16 + 2 * vg;
      pa[2 * vg + 0] = Plds[row][kA + 0];
      pa[2 * vg + 1] = Plds[row][kA + 1];
    }
    // o += P @ V, then issue next-step V fragment loads.
    #pragma unroll
    for (int n = 0; n < 4; ++n)
      o[n] = __builtin_amdgcn_wmma_f32_16x16x32_f16(false, pa, false, vf[n], (short)0, o[n], false, false);
    if (more) {
      #pragma unroll
      for (int n = 0; n < 4; ++n)
        vf[n] = load16h(vbase + (size_t)(n * 16 + row) * SEQ + s1 + hi * 16);
    }
    __syncthreads();  // Plds reused next step
  }

  // Normalize and store output [B, SEQ, HS] f32.
  #pragma unroll
  for (int r = 0; r < 8; ++r) {
    const int M = hi ? (8 + r) : r;
    const float inv = 1.0f / l_i[r];
    const size_t base = ((size_t)b * SEQ + t0 + M) * HS;
    #pragma unroll
    for (int n = 0; n < 4; ++n) out[base + n * 16 + row] = o[n][r] * inv;
  }
}

// ---------------------------------------------------------------------------
extern "C" void kernel_launch(void* const* d_in, const int* in_sizes, int n_in,
                              void* d_out, int out_size, void* d_ws, size_t ws_size,
                              hipStream_t stream) {
  const float* x  = (const float*)d_in[0];
  const float* Wq = (const float*)d_in[1];
  const float* Wk = (const float*)d_in[2];
  const float* Wv = (const float*)d_in[3];
  float* out = (float*)d_out;

  _Float16* ws  = (_Float16*)d_ws;
  _Float16* WqT = ws;
  _Float16* WkT = WqT + (size_t)HS * NE;
  _Float16* WvT = WkT + (size_t)HS * NE;
  _Float16* qh  = WvT + (size_t)HS * NE;
  _Float16* kh  = qh + (size_t)BATCH * SEQ * HS;
  _Float16* vT  = kh + (size_t)BATCH * SEQ * HS;

  convert_weights<<<(HS * NE + 255) / 256, 256, 0, stream>>>(Wq, Wk, Wv, WqT, WkT, WvT);
  proj_qkv<<<BATCH * (SEQ / 16), 32, 0, stream>>>(x, WqT, WkT, WvT, qh, kh, vT);
  attn<<<BATCH * (SEQ / 16), 32, 0, stream>>>(qh, kh, vT, out);
}
